// GraphTransformer_39393440039566
// MI455X (gfx1250) — matrix operand order repaired and checked by reading, not compiled, not run on hardware
//
#include <hip/hip_runtime.h>
#include <hip/hip_bf16.h>

typedef __attribute__((ext_vector_type(16))) __bf16 v16bf;
typedef __attribute__((ext_vector_type(8)))  float  v8f;
typedef unsigned short ushort_t;

// ---------------------------------------------------------------------------
// helpers
// ---------------------------------------------------------------------------
__device__ __forceinline__ unsigned short f2bf(float f) {
    unsigned b = __float_as_uint(f);
    unsigned r = b + 0x7FFFu + ((b >> 16) & 1u);   // round-to-nearest-even
    return (unsigned short)(r >> 16);
}
// order-preserving float <-> uint encoding (for atomicMax segment-max)
__device__ __forceinline__ unsigned fenc(float f) {
    unsigned b = __float_as_uint(f);
    return (b & 0x80000000u) ? ~b : (b | 0x80000000u);
}
__device__ __forceinline__ float fdec(unsigned u) {
    unsigned b = (u & 0x80000000u) ? (u & 0x7FFFFFFFu) : ~u;
    return __uint_as_float(b);
}

union BfPack { v16bf v; uint4 u[2]; };

// ---------------------------------------------------------------------------
// f32 -> bf16 conversion (optionally fused ReLU)
// ---------------------------------------------------------------------------
__global__ void k_f32_to_bf16(const float* __restrict__ in,
                              ushort_t* __restrict__ out, int n, int relu) {
    int i = blockIdx.x * blockDim.x + threadIdx.x;
    if (i < n) {
        float f = in[i];
        if (relu) f = fmaxf(f, 0.0f);
        out[i] = f2bf(f);
    }
}

// ---------------------------------------------------------------------------
// init segment-max / denom buffers
// ---------------------------------------------------------------------------
__global__ void k_init_seg(unsigned* __restrict__ maxb,
                           float* __restrict__ den, int n) {
    int i = blockIdx.x * blockDim.x + threadIdx.x;
    if (i < n) {
        maxb[i] = fenc(-INFINITY);
        den[i]  = 0.0f;
    }
}

// ---------------------------------------------------------------------------
// fused Q/K/V/S GEMM: per block, 4 waves; wave w computes x_tile @ W_w^T + b_w
// 16 rows x 64 cols per wave via v_wmma_f32_16x16x32_bf16 (2 WMMAs / 16x16).
// qkv = contiguous [3][N][64]; sout gets the skip-connection (x@Ws^T + bs).
// ---------------------------------------------------------------------------
__global__ void k_gemm_qkvs(const ushort_t* __restrict__ xbf,
                            const ushort_t* __restrict__ wbf,   // 4 mats, each 64x64 row-major
                            const float* __restrict__ bq, const float* __restrict__ bk,
                            const float* __restrict__ bv, const float* __restrict__ bs,
                            float* __restrict__ qkv, float* __restrict__ sout, int N) {
    const int lane = threadIdx.x & 31;
    const int widx = threadIdx.x >> 5;          // 0..3 -> Q,K,V,S
    const int m0   = blockIdx.x * 16;
    const bool full = (m0 + 16 <= N);           // uniform: whole tile in range

    // ---- A tile: 16x64 bf16 of x, per ISA 16-bit A layout ------------------
    // lanes 0-15: M=lane, chunks K=[kb+0..7] and K=[kb+16..23]
    // lanes16-31: M=lane-16, chunks K=[kb+8..15] and K=[kb+24..31]
    int arow = m0 + (lane & 15);
    if (!full && arow >= N) arow = N - 1;       // clamp only on tail tile
    const ushort_t* abase = xbf + (size_t)arow * 64;
    const int ka = (lane & 16) >> 1;            // 0 or 8

    BfPack A0, A1;
    A0.u[0] = *(const uint4*)(abase + 0  + ka);
    A0.u[1] = *(const uint4*)(abase + 16 + ka);
    A1.u[0] = *(const uint4*)(abase + 32 + ka);
    A1.u[1] = *(const uint4*)(abase + 48 + ka);

    const ushort_t* wmat = wbf + (size_t)widx * 64 * 64;
    const float* bias = (widx == 0) ? bq : (widx == 1) ? bk : (widx == 2) ? bv : bs;
    float* dst = (widx < 3) ? (qkv + (size_t)widx * N * 64) : sout;

    const int nloc = lane & 15;
    const int kb_half = (lane & 16) >> 1;       // 0 or 8 (B: lanes16-31 hold K=8..15,24..31)
    const int rbase = m0 + ((lane & 16) ? 8 : 0);

    for (int n0 = 0; n0 < 64; n0 += 16) {
        // ---- B tiles: 32x16 bf16, lane n holds contiguous K chunk of W row -
        const ushort_t* brow = wmat + (size_t)(n0 + nloc) * 64;
        BfPack B0, B1;
        B0.u[0] = *(const uint4*)(brow + 0  + kb_half);
        B0.u[1] = *(const uint4*)(brow + 16 + kb_half);
        B1.u[0] = *(const uint4*)(brow + 32 + kb_half);
        B1.u[1] = *(const uint4*)(brow + 48 + kb_half);

        v8f acc = {};
        acc = __builtin_amdgcn_wmma_f32_16x16x32_bf16(false, A0.v, false, B0.v,
                                                      (short)0, acc, false, false);
        acc = __builtin_amdgcn_wmma_f32_16x16x32_bf16(false, A1.v, false, B1.v,
                                                      (short)0, acc, false, false);

        const float bval = bias[n0 + nloc];
        float* dcol = dst + (size_t)rbase * 64 + n0 + nloc;
        if (full) {                              // uniform fast path: no guards
            #pragma unroll
            for (int i = 0; i < 8; ++i)
                dcol[(size_t)i * 64] = acc[i] + bval;
        } else {
            #pragma unroll
            for (int i = 0; i < 8; ++i)
                if (rbase + i < N) dcol[(size_t)i * 64] = acc[i] + bval;
        }
    }
}

// ---------------------------------------------------------------------------
// edge score: 16 lanes per edge, float4 gathers, half-wave reduction,
// ordered-uint atomicMax for segment max over dst
// ---------------------------------------------------------------------------
__global__ void k_edge_score(const float* __restrict__ q, const float* __restrict__ k,
                             const int* __restrict__ ei, float* __restrict__ score,
                             unsigned* __restrict__ maxb, int E) {
    const int e = blockIdx.x * 16 + (threadIdx.x >> 4);
    const int j = threadIdx.x & 15;
    if (e >= E) return;
    const int src = ei[e];
    const int dstn = ei[E + e];
    const float4 qv = *(const float4*)(q + (size_t)dstn * 64 + j * 4);
    const float4 kv = *(const float4*)(k + (size_t)src  * 64 + j * 4);
    float p = fmaf(qv.x, kv.x, fmaf(qv.y, kv.y, fmaf(qv.z, kv.z, qv.w * kv.w)));
    #pragma unroll
    for (int m = 8; m >= 1; m >>= 1) p += __shfl_xor(p, m, 16);
    if (j == 0) {
        const float s = p * 0.125f;             // 1/sqrt(64)
        score[e] = s;
        atomicMax(maxb + dstn, fenc(s));
    }
}

// ---------------------------------------------------------------------------
// e = exp(score - max[dst]); denom[dst] += e  (score overwritten in place)
// ---------------------------------------------------------------------------
__global__ void k_edge_exp(float* __restrict__ score,
                           const unsigned* __restrict__ maxb,
                           float* __restrict__ den,
                           const int* __restrict__ ei, int E) {
    const int e = blockIdx.x * blockDim.x + threadIdx.x;
    if (e >= E) return;
    const int dstn = ei[E + e];
    const float m = fdec(maxb[dstn]);
    const float ex = __expf(score[e] - m);
    score[e] = ex;
    atomicAdd(den + dstn, ex);
}

// ---------------------------------------------------------------------------
// out[dst] += (e/denom[dst]) * v[src] ; out pre-initialized with x@Ws^T + bs
// ---------------------------------------------------------------------------
__global__ void k_edge_scatter(const float* __restrict__ score,
                               const float* __restrict__ den,
                               const float* __restrict__ v,
                               const int* __restrict__ ei,
                               float* __restrict__ out, int E) {
    const int e = blockIdx.x * 16 + (threadIdx.x >> 4);
    const int j = threadIdx.x & 15;
    if (e >= E) return;
    const int src = ei[e];
    const int dstn = ei[E + e];
    const float alpha = score[e] / den[dstn];
    const float4 vv = *(const float4*)(v + (size_t)src * 64 + j * 4);
    float* o = out + (size_t)dstn * 64 + j * 4;
    atomicAdd(o + 0, alpha * vv.x);
    atomicAdd(o + 1, alpha * vv.y);
    atomicAdd(o + 2, alpha * vv.z);
    atomicAdd(o + 3, alpha * vv.w);
}

// ---------------------------------------------------------------------------
// host
// ---------------------------------------------------------------------------
extern "C" void kernel_launch(void* const* d_in, const int* in_sizes, int n_in,
                              void* d_out, int out_size, void* d_ws, size_t ws_size,
                              hipStream_t stream) {
    const float* x  = (const float*)d_in[0];
    const int*   ei = (const int*)d_in[1];
    const int N = in_sizes[0] / 64;
    const int E = in_sizes[1] / 2;

    char* ws = (char*)d_ws;
    size_t off = 0;
    auto carve = [&](size_t bytes) -> void* {
        void* p = ws + off;
        off = (off + bytes + 255) & ~(size_t)255;
        return p;
    };
    ushort_t* xbf  = (ushort_t*)carve((size_t)N * 64 * 2);
    ushort_t* wbf  = (ushort_t*)carve((size_t)4 * 64 * 64 * 2);
    float*    qkv  = (float*)carve((size_t)3 * N * 64 * 4);   // q | k | v contiguous
    float*    sc   = (float*)carve((size_t)E * 4);
    unsigned* maxb = (unsigned*)carve((size_t)N * 4);
    float*    den  = (float*)carve((size_t)N * 4);
    float*    h    = (float*)carve((size_t)N * 64 * 4);

    float* qb = qkv;
    float* kb = qkv + (size_t)N * 64;
    float* vb = qkv + (size_t)2 * N * 64;

    auto run_layer = [&](const float* in, int relu,
                         const float* Wq, const float* bq,
                         const float* Wk, const float* bk,
                         const float* Wv, const float* bv,
                         const float* Wsk, const float* bs,
                         float* outp) {
        k_f32_to_bf16<<<(N * 64 + 255) / 256, 256, 0, stream>>>(in, xbf, N * 64, relu);
        k_f32_to_bf16<<<16, 256, 0, stream>>>(Wq,  wbf + 0 * 4096, 4096, 0);
        k_f32_to_bf16<<<16, 256, 0, stream>>>(Wk,  wbf + 1 * 4096, 4096, 0);
        k_f32_to_bf16<<<16, 256, 0, stream>>>(Wv,  wbf + 2 * 4096, 4096, 0);
        k_f32_to_bf16<<<16, 256, 0, stream>>>(Wsk, wbf + 3 * 4096, 4096, 0);
        k_init_seg<<<(N + 255) / 256, 256, 0, stream>>>(maxb, den, N);
        k_gemm_qkvs<<<(N + 15) / 16, 128, 0, stream>>>(xbf, wbf, bq, bk, bv, bs,
                                                       qkv, outp, N);
        k_edge_score<<<(E + 15) / 16, 256, 0, stream>>>(qb, kb, ei, sc, maxb, E);
        k_edge_exp<<<(E + 255) / 256, 256, 0, stream>>>(sc, maxb, den, ei, E);
        k_edge_scatter<<<(E + 15) / 16, 256, 0, stream>>>(sc, den, vb, ei, outp, E);
    };

    // layer 1: x -> h
    run_layer(x, /*relu=*/0,
              (const float*)d_in[2], (const float*)d_in[3],
              (const float*)d_in[4], (const float*)d_in[5],
              (const float*)d_in[6], (const float*)d_in[7],
              (const float*)d_in[8], (const float*)d_in[9],
              h);
    // layer 2: relu(h) -> d_out
    run_layer(h, /*relu=*/1,
              (const float*)d_in[10], (const float*)d_in[11],
              (const float*)d_in[12], (const float*)d_in[13],
              (const float*)d_in[14], (const float*)d_in[15],
              (const float*)d_in[16], (const float*)d_in[17],
              (float*)d_out);
}
